// LayerNormLSTM_54889682043588
// MI455X (gfx1250) — compile-verified, matrix operand order checked by reading, not run on hardware
//
#include <hip/hip_runtime.h>

// Problem constants (from reference)
#define Tt    256
#define Bb    128
#define Hh    1024
#define Ll    2
#define FourH 4096
#define TN    256      // N tiles of 16 (4096/16)
#define TNP   128      // N tile *pairs* (4096/32)
#define TK    32       // K tiles of 32 (1024/32)
#define EPSv  1e-5f
#define APAD  1032     // padded LDS row stride (elements) to spread banks

typedef __attribute__((ext_vector_type(16))) __bf16        v16bf;
typedef __attribute__((ext_vector_type(8)))  float         v8f;
typedef __attribute__((ext_vector_type(4)))  unsigned int  v4u;

union BFrag { v16bf v; v4u q[2]; unsigned short u[16]; };

__device__ __forceinline__ unsigned short f2bf(float f) {
    union { float f; unsigned int u; } c; c.f = f;
    unsigned int r = c.u + 0x7FFFu + ((c.u >> 16) & 1u); // round-to-nearest-even
    return (unsigned short)(r >> 16);
}

__device__ __forceinline__ float sigm(float x) { return 1.0f / (1.0f + __expf(-x)); }

// ---------------------------------------------------------------------------
// One-time weight repack: f32 [L][4H][H] -> bf16 WMMA B-fragment tiles.
// Packed layout: [l][which][tile_pair][tile_k][sub(2)][lane][16]. A wave's two
// N-tiles at the same K-step are ADJACENT (2 KB chunk), so the GEMM loop loads
// all four b128 fragments from one scalar base + static offsets.
// ---------------------------------------------------------------------------
__global__ __launch_bounds__(256) void pack_w(const float* __restrict__ w_ih,
                                              const float* __restrict__ w_hh,
                                              unsigned short* __restrict__ packed) {
    long long idx = (long long)blockIdx.x * 256 + threadIdx.x;
    const long long total = (long long)Ll * 2 * TNP * TK * 2 * 32 * 16;
    if (idx >= total) return;
    int j = (int)(idx & 15);
    long long r = idx >> 4;
    int lane = (int)(r & 31); r >>= 5;
    int sub  = (int)(r & 1);  r >>= 1;
    int tk = (int)(r % TK);   r /= TK;
    int tnp = (int)(r % TNP); r /= TNP;
    int which = (int)(r & 1); r >>= 1;
    int l = (int)r;
    int n = (tnp * 2 + sub) * 16 + (lane & 15);
    int khalf = lane >> 4;
    int k = tk * 32 + ((j < 8) ? (khalf * 8 + j) : (16 + khalf * 8 + (j - 8)));
    const float* W = which ? w_hh : w_ih;
    float val = W[((long long)l * FourH + n) * Hh + k];
    packed[idx] = f2bf(val);
}

__global__ void zero_f32(float* __restrict__ p, long long n) {
    long long i = (long long)blockIdx.x * blockDim.x + threadIdx.x;
    if (i < n) p[i] = 0.0f;
}

__global__ void copy_f32(const float* __restrict__ s, float* __restrict__ d, long long n) {
    long long i = (long long)blockIdx.x * blockDim.x + threadIdx.x;
    if (i < n) d[i] = s[i];
}

// ---------------------------------------------------------------------------
// Gate pre-activation GEMM for one (t, layer):
//   G[which][128][4096] = A @ W^T + b; which=0: A=layer input, which=1: A=h_prev
// Block = 256 threads = 8 wave32 waves. The block's full 16x1024 A stripe is
// staged f32->bf16 into LDS once (single barrier); the K loop is barrier-free:
// 2x ds_load_b128 (A frag, bank-padded) + 4x global b128 (paired B frags from
// one scalarized base) + 2x WMMA. Each wave owns a 16x32 output slab.
// ---------------------------------------------------------------------------
__global__ __launch_bounds__(256) void gemm_gates(
    const float* __restrict__ Ax, const float* __restrict__ Ah,
    const unsigned short* __restrict__ packedL,   // this layer's packed weights
    const float* __restrict__ b_ih_l, const float* __restrict__ b_hh_l,
    float* __restrict__ G)
{
    const int which  = blockIdx.z;
    const float* A    = which ? Ah : Ax;
    const float* bias = which ? b_hh_l : b_ih_l;

    const int m_tile = blockIdx.y;                 // 0..7   (M rows /16)
    const int wave   = threadIdx.x >> 5;           // 0..7
    const int lane   = threadIdx.x & 31;
    // wave-uniform by construction; force into an SGPR for saddr addressing
    const int n_pair = __builtin_amdgcn_readfirstlane(blockIdx.x * 8 + wave);

    __shared__ __align__(16) unsigned short As[16 * APAD]; // full-K A stripe, bf16

    // ---- prologue: stage A[16][1024] (f32 -> bf16) into LDS once ----
    {
        // 16 consecutive rows of row-major [B][H] are contiguous: coalesced float4.
        const float4* Af = (const float4*)(A + (long long)m_tile * 16 * Hh);
        for (int it = threadIdx.x; it < (16 * Hh) / 4; it += 256) {
            float4 v = Af[it];
            int e   = it * 4;
            int row = e >> 10;          // /1024
            int col = e & (Hh - 1);
            unsigned short* dst = &As[row * APAD + col];
            dst[0] = f2bf(v.x); dst[1] = f2bf(v.y);
            dst[2] = f2bf(v.z); dst[3] = f2bf(v.w);
        }
    }
    __syncthreads();

    const int e0 = (lane >> 4) * 8;    // k-half select (lanes 16-31 -> +8)
    const int m  = lane & 15;

    // Scalar base for this wave's B pair stream: [which][n_pair] chunk start.
    // Each K-step chunk is 2 tiles * 512 elem = 1024 elements (2 KB).
    const v4u* Bq = (const v4u*)(packedL
                    + ((size_t)which * TNP + n_pair) * (size_t)TK * 1024);

    v8f acc0 = {}, acc1 = {};

#pragma unroll 4
    for (int kk = 0; kk < TK; ++kk) {
        // A fragment from LDS (16-bit 16x32 A layout), bank-padded stride
        BFrag af;
        af.q[0] = *(const v4u*)&As[m * APAD + kk * 32 + e0];
        af.q[1] = *(const v4u*)&As[m * APAD + kk * 32 + 16 + e0];

        // Paired B fragments: one base, static offsets {0,16,1024,1040} bytes
        const v4u* bp = Bq + kk * 128 + lane * 2;
        BFrag bf0, bf1;
        bf0.q[0] = bp[0];  bf0.q[1] = bp[1];
        bf1.q[0] = bp[64]; bf1.q[1] = bp[65];

        acc0 = __builtin_amdgcn_wmma_f32_16x16x32_bf16(
                   false, af.v, false, bf0.v, (short)0, acc0, false, false);
        acc1 = __builtin_amdgcn_wmma_f32_16x16x32_bf16(
                   false, af.v, false, bf1.v, (short)0, acc1, false, false);
    }

    // Epilogue: C layout (lane n = lane&15; vgpr v -> row v + 8*(lane>=16)) + bias
    const int n_loc  = lane & 15;
    const int mbase  = (lane >> 4) * 8;
    const int aRow0  = m_tile * 16;
    const int ng0 = (n_pair * 2) * 16 + n_loc;
    const int ng1 = (n_pair * 2 + 1) * 16 + n_loc;
    const float bv0 = bias[ng0];
    const float bv1 = bias[ng1];
#pragma unroll
    for (int v = 0; v < 8; ++v) {
        int m_glob = aRow0 + mbase + v;
        long long rowoff = ((long long)which * Bb + m_glob) * FourH;
        G[rowoff + ng0] = acc0[v] + bv0;
        G[rowoff + ng1] = acc1[v] + bv1;
    }
}

// ---------------------------------------------------------------------------
// Pointwise LayerNorm-LSTM cell per batch row: two 4096-wide LNs, gates,
// c-update, 1024-wide LN on c, h output. One block per row, 256 threads.
// G row elements are cached in registers during the reduction pass so the
// gate pass issues no further G loads.
// ---------------------------------------------------------------------------
__device__ __forceinline__ void block_reduce2(float& a, float& b,
                                              float* s1, float* s2) {
    int tid = threadIdx.x;
    s1[tid] = a; s2[tid] = b;
    __syncthreads();
    for (int off = 128; off > 0; off >>= 1) {
        if (tid < off) { s1[tid] += s1[tid + off]; s2[tid] += s2[tid + off]; }
        __syncthreads();
    }
    a = s1[0]; b = s2[0];
    __syncthreads();
}

__global__ __launch_bounds__(256) void ln_lstm_pointwise(
    const float* __restrict__ G,
    const float* __restrict__ g_ih_l, const float* __restrict__ be_ih_l,
    const float* __restrict__ g_hh_l, const float* __restrict__ be_hh_l,
    const float* __restrict__ g_ho_l, const float* __restrict__ be_ho_l,
    float* __restrict__ hRow, float* __restrict__ cRow,
    float* __restrict__ outT, int writeOut)
{
    const int b   = blockIdx.x;
    const int tid = threadIdx.x;
    const float* Gi = G + (long long)b * FourH;
    const float* Gh = G + ((long long)Bb + b) * FourH;

    __shared__ float s1[256], s2[256];

    float gir[16], ghr[16];
    float si = 0, qi = 0, sh = 0, qh = 0;
#pragma unroll
    for (int it = 0; it < 16; ++it) {
        int x = tid + it * 256;
        float a = Gi[x]; gir[it] = a; si += a; qi += a * a;
        float c = Gh[x]; ghr[it] = c; sh += c; qh += c * c;
    }
    block_reduce2(si, qi, s1, s2);
    block_reduce2(sh, qh, s1, s2);
    const float mi = si * (1.0f / FourH);
    const float ri = rsqrtf(qi * (1.0f / FourH) - mi * mi + EPSv);
    const float mh = sh * (1.0f / FourH);
    const float rh = rsqrtf(qh * (1.0f / FourH) - mh * mh + EPSv);

    float cnew[4], og[4];
    float sc = 0, qc = 0;
#pragma unroll
    for (int it = 0; it < 4; ++it) {
        int j = tid + it * 256;
        // cached indices: x = j, H+j, 2H+j, 3H+j  ->  it, it+4, it+8, it+12
        float gi = (gir[it]      - mi) * ri * g_ih_l[j]          + be_ih_l[j]
                 + (ghr[it]      - mh) * rh * g_hh_l[j]          + be_hh_l[j];
        float gf = (gir[it + 4]  - mi) * ri * g_ih_l[Hh + j]     + be_ih_l[Hh + j]
                 + (ghr[it + 4]  - mh) * rh * g_hh_l[Hh + j]     + be_hh_l[Hh + j];
        float go = (gir[it + 8]  - mi) * ri * g_ih_l[2 * Hh + j] + be_ih_l[2 * Hh + j]
                 + (ghr[it + 8]  - mh) * rh * g_hh_l[2 * Hh + j] + be_hh_l[2 * Hh + j];
        float gg = (gir[it + 12] - mi) * ri * g_ih_l[3 * Hh + j] + be_ih_l[3 * Hh + j]
                 + (ghr[it + 12] - mh) * rh * g_hh_l[3 * Hh + j] + be_hh_l[3 * Hh + j];
        float ig = sigm(gi), fg = sigm(gf), oo = sigm(go), g2 = tanhf(gg);
        float c0 = cRow[(long long)b * Hh + j];
        float cn = fg * c0 + ig * g2;
        cRow[(long long)b * Hh + j] = cn;
        cnew[it] = cn; og[it] = oo;
        sc += cn; qc += cn * cn;
    }
    block_reduce2(sc, qc, s1, s2);
    const float mc = sc * (1.0f / Hh);
    const float rc = rsqrtf(qc * (1.0f / Hh) - mc * mc + EPSv);
#pragma unroll
    for (int it = 0; it < 4; ++it) {
        int j = tid + it * 256;
        float hv = og[it] * tanhf((cnew[it] - mc) * rc * g_ho_l[j] + be_ho_l[j]);
        hRow[(long long)b * Hh + j] = hv;
        if (writeOut) outT[(long long)b * Hh + j] = hv;
    }
}

// ---------------------------------------------------------------------------
extern "C" void kernel_launch(void* const* d_in, const int* in_sizes, int n_in,
                              void* d_out, int out_size, void* d_ws, size_t ws_size,
                              hipStream_t stream) {
    (void)in_sizes; (void)n_in; (void)out_size; (void)ws_size;
    const float* x     = (const float*)d_in[0];   // [T,B,H]
    const float* w_ih  = (const float*)d_in[1];   // [L,4H,H]
    const float* w_hh  = (const float*)d_in[2];   // [L,4H,H]
    const float* b_ih  = (const float*)d_in[3];   // [L,4H]
    const float* b_hh  = (const float*)d_in[4];   // [L,4H]
    const float* g_ih  = (const float*)d_in[5];   // [L,4H]
    const float* be_ih = (const float*)d_in[6];
    const float* g_hh  = (const float*)d_in[7];
    const float* be_hh = (const float*)d_in[8];
    const float* g_ho  = (const float*)d_in[9];   // [L,H]
    const float* be_ho = (const float*)d_in[10];
    float* out = (float*)d_out;                   // out[T,B,H] ++ hy[L,B,H] ++ cy[L,B,H]

    // Workspace layout
    char* ws = (char*)d_ws;
    unsigned short* packed = (unsigned short*)ws;                  // L*2*TNP*TK*1024 bf16 (33.5 MB)
    const size_t packedElems = (size_t)Ll * 2 * TNP * TK * 1024;
    float* G      = (float*)(ws + packedElems * sizeof(unsigned short)); // [2][B][4H] (4 MB)
    float* hState = (float*)((char*)G + (size_t)2 * Bb * FourH * sizeof(float)); // [L][B][H]
    float* cState = hState + (size_t)Ll * Bb * Hh;                        // [L][B][H]

    // 1) pack weights (f32 -> bf16 WMMA fragment layout); L2-resident thereafter
    {
        long long total = (long long)Ll * 2 * TNP * TK * 2 * 32 * 16;
        pack_w<<<dim3((unsigned)((total + 255) / 256)), dim3(256), 0, stream>>>(w_ih, w_hh, packed);
    }
    // 2) zero h/c state (contiguous)
    {
        long long n = (long long)2 * Ll * Bb * Hh;
        zero_f32<<<dim3((unsigned)((n + 255) / 256)), dim3(256), 0, stream>>>(hState, n);
    }

    // 3) serial recurrence: per (t, layer) one WMMA GEMM kernel + one pointwise kernel
    dim3 ggrid(TN / 16, Bb / 16, 2);   // (16, 8, 2); 8 waves x 2 N-tiles each
    for (int t = 0; t < Tt; ++t) {
        for (int l = 0; l < Ll; ++l) {
            const float* Ax = (l == 0) ? (x + (long long)t * Bb * Hh)
                                       : (hState + 0);                 // layer-0 h just updated
            const float* Ah = hState + (long long)l * Bb * Hh;         // this layer's h(t-1)
            const unsigned short* packedL = packed + (size_t)l * 2 * TNP * TK * 1024;
            gemm_gates<<<ggrid, dim3(256), 0, stream>>>(
                Ax, Ah, packedL,
                b_ih + (long long)l * FourH, b_hh + (long long)l * FourH, G);
            ln_lstm_pointwise<<<dim3(Bb), dim3(256), 0, stream>>>(
                G,
                g_ih + (long long)l * FourH, be_ih + (long long)l * FourH,
                g_hh + (long long)l * FourH, be_hh + (long long)l * FourH,
                g_ho + (long long)l * Hh,   be_ho + (long long)l * Hh,
                hState + (long long)l * Bb * Hh, cState + (long long)l * Bb * Hh,
                out + (long long)t * Bb * Hh, (l == Ll - 1) ? 1 : 0);
        }
    }

    // 4) final states: hy then cy (hState/cState are contiguous in ws)
    {
        long long n = (long long)2 * Ll * Bb * Hh;
        copy_f32<<<dim3((unsigned)((n + 255) / 256)), dim3(256), 0, stream>>>(
            hState, out + (long long)Tt * Bb * Hh, n);
    }
}
